// GraphConvolution_68805376082138
// MI455X (gfx1250) — matrix operand order repaired and checked by reading, not compiled, not run on hardware
//
#include <hip/hip_runtime.h>

// ---------------------------------------------------------------------------
// GCN layer for MI455X (gfx1250, wave32):
//   h   = x @ W                         -> f16 WMMA GEMM, f32 accumulation
//   agg = scatter_add(h[src]*ew, dst)   -> global_atomic_add_f32 (L2-resident h)
//   out = relu(agg + bias)
// d_ws holds h: N*F_OUT*4 = 25.6 MB.
// ---------------------------------------------------------------------------

typedef __attribute__((ext_vector_type(16))) _Float16     v16h;
typedef __attribute__((ext_vector_type(8)))  float        v8f;
typedef __attribute__((ext_vector_type(4)))  float        f32x4;
typedef __attribute__((ext_vector_type(4)))  unsigned int u32x4;
typedef __attribute__((ext_vector_type(2)))  unsigned int u32x2;

#define F_IN   512
#define F_OUT  128
#define TILE_M 128
#define TILE_K 32
#define STR    40   // LDS row stride in f16 (32 data + 8 pad); 80B rows, 16B aligned

union AFrag { v16h v; u32x4 u[2]; };
union H4    { _Float16 h[4]; u32x2 u; };

// ---------------------------------------------------------------------------
// Kernel 1: h = x @ W (M=nRows, K=512, N=128). Block = 256 threads (8 wave32),
// tile 128(M) x 128(N), K step 32, software-pipelined one stage deep:
//   loop: [barrier] issue global loads tile k -> regs | WMMA tile k-1 from LDS
//         [barrier] cvt+store regs -> LDS
// ---------------------------------------------------------------------------
__global__ __launch_bounds__(256, 2)
void gemm_xw_wmma(const float* __restrict__ x, const float* __restrict__ w,
                  float* __restrict__ h, int nRows) {
  __shared__ __attribute__((aligned(16))) _Float16 As[TILE_M * STR];  // x tile  [row][k]
  __shared__ __attribute__((aligned(16))) _Float16 Bs[F_OUT * STR];   // W^T tile[n][k]

  const int tid  = threadIdx.x;
  const int wave = tid >> 5;
  const int lane = tid & 31;
  const int half = lane >> 4;
  const int l16  = lane & 15;
  const size_t rowBase = (size_t)blockIdx.x * TILE_M;

  // ---- precomputed staging addresses (loop-invariant) ----------------------
  const float* ap[4];  int aso[4];           // A: thread -> 4x f32x4 (row r, cols c..c+3)
  const float* bp[4];  int bso[4];           // B: thread -> (n, kk..kk+3), transposed store
#pragma unroll
  for (int i = 0; i < 4; ++i) {
    int e = (tid + i * 256) * 4;             // element in 128x32 tile
    int r = e >> 5, c = e & 31;
    size_t gr = rowBase + (size_t)r;
    if (gr >= (size_t)nRows) gr = (size_t)nRows - 1;   // clamp (tail block only)
    ap[i]  = x + gr * F_IN + c;
    aso[i] = r * STR + c;

    int idx = tid + i * 256;                 // 0..1023 over (n, kk/4)
    int n  = idx & 127;
    int kk = (idx >> 7) * 4;
    bp[i]  = w + (size_t)kk * F_OUT + n;
    bso[i] = n * STR + kk;
  }

  // ---- precomputed fragment LDS pointers ----------------------------------
  const _Float16* aF = &As[(wave * 16 + l16) * STR + half * 16]; // K runs 0-15 | 16-31
  const _Float16* bF = &Bs[l16 * STR + half * 16];               // K runs 0-15 | 16-31

  v8f acc[8];
#pragma unroll
  for (int j = 0; j < 8; ++j) acc[j] = v8f{};

  f32x4 Ar[4], Br[4];

  // ---- prologue: load + stage tile 0 --------------------------------------
#pragma unroll
  for (int i = 0; i < 4; ++i) {
    Ar[i] = *(const f32x4*)(ap[i]);
    const float* p = bp[i];
    Br[i].x = p[0]; Br[i].y = p[F_OUT]; Br[i].z = p[2 * F_OUT]; Br[i].w = p[3 * F_OUT];
  }
#pragma unroll
  for (int i = 0; i < 4; ++i) {
    H4 ta; ta.h[0] = (_Float16)Ar[i].x; ta.h[1] = (_Float16)Ar[i].y;
           ta.h[2] = (_Float16)Ar[i].z; ta.h[3] = (_Float16)Ar[i].w;
    *(u32x2*)&As[aso[i]] = ta.u;
    H4 tb; tb.h[0] = (_Float16)Br[i].x; tb.h[1] = (_Float16)Br[i].y;
           tb.h[2] = (_Float16)Br[i].z; tb.h[3] = (_Float16)Br[i].w;
    *(u32x2*)&Bs[bso[i]] = tb.u;
  }

  // ---- pipelined main loop -------------------------------------------------
  for (int k0 = TILE_K; k0 < F_IN; k0 += TILE_K) {
    __syncthreads();                         // LDS tile (k0-32) visible

    // issue next tile's global loads (consumed after 2nd barrier)
#pragma unroll
    for (int i = 0; i < 4; ++i) {
      Ar[i] = *(const f32x4*)(ap[i] + k0);
      const float* p = bp[i] + (size_t)k0 * F_OUT;
      Br[i].x = p[0]; Br[i].y = p[F_OUT]; Br[i].z = p[2 * F_OUT]; Br[i].w = p[3 * F_OUT];
    }

    // compute tile (k0-32) from LDS
    {
      AFrag A;
      A.u[0] = *(const u32x4*)(aF);
      A.u[1] = *(const u32x4*)(aF + 8);
#pragma unroll
      for (int j = 0; j < 8; ++j) {
        AFrag B;
        B.u[0] = *(const u32x4*)(bF + j * 16 * STR);
        B.u[1] = *(const u32x4*)(bF + j * 16 * STR + 8);
        acc[j] = __builtin_amdgcn_wmma_f32_16x16x32_f16(
            false, A.v, false, B.v, (short)0, acc[j], false, false);
      }
    }

    __syncthreads();                         // everyone done reading LDS

    // stage tile k0 into LDS
#pragma unroll
    for (int i = 0; i < 4; ++i) {
      H4 ta; ta.h[0] = (_Float16)Ar[i].x; ta.h[1] = (_Float16)Ar[i].y;
             ta.h[2] = (_Float16)Ar[i].z; ta.h[3] = (_Float16)Ar[i].w;
      *(u32x2*)&As[aso[i]] = ta.u;
      H4 tb; tb.h[0] = (_Float16)Br[i].x; tb.h[1] = (_Float16)Br[i].y;
             tb.h[2] = (_Float16)Br[i].z; tb.h[3] = (_Float16)Br[i].w;
      *(u32x2*)&Bs[bso[i]] = tb.u;
    }
  }

  // ---- epilogue: last tile's WMMAs ----------------------------------------
  __syncthreads();
  {
    AFrag A;
    A.u[0] = *(const u32x4*)(aF);
    A.u[1] = *(const u32x4*)(aF + 8);
#pragma unroll
    for (int j = 0; j < 8; ++j) {
      AFrag B;
      B.u[0] = *(const u32x4*)(bF + j * 16 * STR);
      B.u[1] = *(const u32x4*)(bF + j * 16 * STR + 8);
      acc[j] = __builtin_amdgcn_wmma_f32_16x16x32_f16(
          false, A.v, false, B.v, (short)0, acc[j], false, false);
    }
  }

  // ---- store: C layout VGPR r -> row r / 8+r by lane-half ------------------
  const size_t mrow = rowBase + (size_t)(wave * 16 + half * 8);
  if (rowBase + TILE_M <= (size_t)nRows) {
    // full tile: unguarded stores, immediate offsets (r*512B)
#pragma unroll
    for (int j = 0; j < 8; ++j) {
      float* p = h + mrow * F_OUT + j * 16 + l16;
#pragma unroll
      for (int r = 0; r < 8; ++r) p[r * F_OUT] = acc[j][r];
    }
  } else {
#pragma unroll
    for (int j = 0; j < 8; ++j) {
      float* p = h + mrow * F_OUT + j * 16 + l16;
#pragma unroll
      for (int r = 0; r < 8; ++r)
        if (mrow + (size_t)r < (size_t)nRows) p[r * F_OUT] = acc[j][r];
    }
  }
}

// ---------------------------------------------------------------------------
// Kernel 2: zero the aggregation buffer (d_out is poisoned by the harness).
// ---------------------------------------------------------------------------
__global__ void zero_f32x4(float* __restrict__ out, int n4) {
  int i = blockIdx.x * blockDim.x + threadIdx.x;
  if (i < n4) ((f32x4*)out)[i] = f32x4{};
}

// ---------------------------------------------------------------------------
// Kernel 3: edge-parallel scatter-add. One wave32 per edge; each lane owns
// 4 consecutive features (32*4 = 128 = F_OUT). h (25.6 MB) stays in the
// 192 MB L2, so the random gather and the f32 atomics both land on-chip.
// ---------------------------------------------------------------------------
__global__ __launch_bounds__(256)
void spmm_scatter(const float* __restrict__ h, const long long* __restrict__ ei,
                  const float* __restrict__ ew, float* __restrict__ out, int E) {
  int gid  = blockIdx.x * blockDim.x + threadIdx.x;
  int e    = gid >> 5;                       // wave-uniform edge id
  if (e >= E) return;
  int lane = gid & 31;
  int src  = (int)ei[e];
  int dst  = (int)ei[(size_t)E + e];
  float wt = ew[e];
  f32x4 v = *(const f32x4*)(h + (size_t)src * F_OUT + lane * 4);
  float* o = out + (size_t)dst * F_OUT + lane * 4;
  atomicAdd(o + 0, v.x * wt);
  atomicAdd(o + 1, v.y * wt);
  atomicAdd(o + 2, v.z * wt);
  atomicAdd(o + 3, v.w * wt);
}

// ---------------------------------------------------------------------------
// Kernel 4: out = relu(out + bias), in place, float4 per thread.
// ---------------------------------------------------------------------------
__global__ void bias_relu(float* __restrict__ out, const float* __restrict__ bias,
                          int n4) {
  int i = blockIdx.x * blockDim.x + threadIdx.x;
  if (i >= n4) return;
  f32x4 v = ((f32x4*)out)[i];
  int col = (i * 4) & (F_OUT - 1);
  f32x4 b = *(const f32x4*)(bias + col);
  v.x = fmaxf(v.x + b.x, 0.0f);
  v.y = fmaxf(v.y + b.y, 0.0f);
  v.z = fmaxf(v.z + b.z, 0.0f);
  v.w = fmaxf(v.w + b.w, 0.0f);
  ((f32x4*)out)[i] = v;
}

extern "C" void kernel_launch(void* const* d_in, const int* in_sizes, int n_in,
                              void* d_out, int out_size, void* d_ws, size_t ws_size,
                              hipStream_t stream) {
  const float*     x    = (const float*)d_in[0];      // [N, 512]  f32
  const long long* ei   = (const long long*)d_in[1];  // [2, E]    i64
  const float*     ew   = (const float*)d_in[2];      // [E]       f32
  const float*     w    = (const float*)d_in[3];      // [512,128] f32
  const float*     bias = (const float*)d_in[4];      // [128]     f32
  float* out = (float*)d_out;                         // [N, 128]  f32
  float* h   = (float*)d_ws;                          // [N, 128]  f32 scratch

  const int E = in_sizes[2];
  const int N = in_sizes[0] / F_IN;
  const int out4 = out_size / 4;

  dim3 blk(256);
  gemm_xw_wmma<<<dim3((N + TILE_M - 1) / TILE_M), blk, 0, stream>>>(x, w, h, N);
  zero_f32x4<<<dim3((out4 + 255) / 256), blk, 0, stream>>>(out, out4);
  spmm_scatter<<<dim3((E + 7) / 8), blk, 0, stream>>>(h, ei, ew, out, E);
  bias_relu<<<dim3((out4 + 255) / 256), blk, 0, stream>>>(out, bias, out4);
}
